// Interstellar_62259845923151
// MI455X (gfx1250) — matrix operand order repaired and checked by reading, not compiled, not run on hardware
//
#include <hip/hip_runtime.h>

typedef __attribute__((ext_vector_type(16))) __bf16 v16bf;
typedef __attribute__((ext_vector_type(8)))  float  v8f;

#define H    256
#define BSZ  8192
#define RS   264          // LDS row stride in bf16 elements (528 B = 33*16 B, conflict-friendly)
#define MT   64           // batch rows per workgroup

// ---------------- WMMA helpers ----------------

__device__ __forceinline__ v8f wmma_bf16(v16bf a, v16bf b, v8f c) {
  return __builtin_amdgcn_wmma_f32_16x16x32_bf16(false, a, false, b, (short)0, c, false, false);
}

// A fragment (16x32 bf16) from LDS row-major buffer.
// lane l: m = l&15, hi = l>>4; elements e=0..7 <- K=k0+hi*8+e, e=8..15 <- K=k0+16+hi*8+(e-8)
__device__ __forceinline__ v16bf lds_a_frag(const __bf16* buf, int row, int k0, int hi) {
  const __bf16* p = buf + row * RS + k0 + hi * 8;
  union { v16bf v; uint4 q[2]; } u;
  u.q[0] = *(const uint4*)(p);
  u.q[1] = *(const uint4*)(p + 16);
  return u.v;
}

// B fragment (32x16 bf16) from transposed bf16 weights Wt[n][k] (row stride K).
// lane l: n-col = n (precomputed per lane), elements e=0..7 -> K = k0 + hi*8 + e, e=8..15 -> K = k0 + 16 + hi*8 + (e-8)
__device__ __forceinline__ v16bf wt_b_frag(const __bf16* wt, int K, int n, int k0, int hi) {
  const __bf16* p = wt + (size_t)n * K + k0 + hi * 8;
  union { v16bf v; uint4 q[2]; } u;
  u.q[0] = *(const uint4*)(p);
  u.q[1] = *(const uint4*)(p + 16);
  return u.v;
}

// A fragment gathered from a global fp32 embedding row, converted to bf16.
__device__ __forceinline__ v16bf gather_a_frag(const float* emb, int row, int k0, int hi) {
  const float* p = emb + (size_t)row * H + k0 + hi * 8;
  float4 f0 = *(const float4*)(p);
  float4 f1 = *(const float4*)(p + 4);
  float4 f2 = *(const float4*)(p + 16);
  float4 f3 = *(const float4*)(p + 20);
  union { v16bf v; __bf16 e[16]; } u;
  u.e[0]  = (__bf16)f0.x; u.e[1]  = (__bf16)f0.y; u.e[2]  = (__bf16)f0.z; u.e[3]  = (__bf16)f0.w;
  u.e[4]  = (__bf16)f1.x; u.e[5]  = (__bf16)f1.y; u.e[6]  = (__bf16)f1.z; u.e[7]  = (__bf16)f1.w;
  u.e[8]  = (__bf16)f2.x; u.e[9]  = (__bf16)f2.y; u.e[10] = (__bf16)f2.z; u.e[11] = (__bf16)f2.w;
  u.e[12] = (__bf16)f3.x; u.e[13] = (__bf16)f3.y; u.e[14] = (__bf16)f3.z; u.e[15] = (__bf16)f3.w;
  return u.v;
}

// Store a 16x16 f32 C fragment into LDS as bf16. C layout: VGPR r -> m = r + 8*hi, col n.
__device__ __forceinline__ void store_c_bf16(__bf16* buf, int rstrip, int n, int hi, v8f acc) {
#pragma unroll
  for (int r = 0; r < 8; ++r)
    buf[(rstrip + r + 8 * hi) * RS + n] = (__bf16)acc[r];
}

// ---------------- fused stage helpers ----------------

// X = Ax @ WXt (no bias) -> dstX ; Y = Ay @ WYt + biasY -> dstY    (both K=256)
__device__ __forceinline__ void dual_gemm_to_lds(
    const v16bf* aX, const v16bf* aY,
    const __bf16* WXt, const __bf16* WYt, const float* biasY,
    __bf16* dstX, __bf16* dstY,
    int rstrip, int ntbase, int l, int hi)
{
#pragma unroll 1
  for (int nt = 0; nt < 8; ++nt) {
    int n = (ntbase + nt) * 16 + (l & 15);
    v8f accX = {};
    float bv = biasY[n];
    v8f accY = {bv, bv, bv, bv, bv, bv, bv, bv};
#pragma unroll
    for (int k = 0; k < 8; ++k) {
      accX = wmma_bf16(aX[k], wt_b_frag(WXt, 256, n, k * 32, hi), accX);
      accY = wmma_bf16(aY[k], wt_b_frag(WYt, 256, n, k * 32, hi), accY);
    }
    store_c_bf16(dstX, rstrip, n, hi, accX);
    store_c_bf16(dstY, rstrip, n, hi, accY);
  }
}

// g = sigmoid([x|h'] @ Wgt + bg); res = g*h' + (1-g)*x; act(res) -> bufDst   (K=512)
__device__ __forceinline__ void gate_stage(
    const __bf16* bufX, const __bf16* bufY, __bf16* bufDst,
    const __bf16* Wgt, const float* bg,
    int rowA, int rstrip, int ntbase, int l, int hi, bool isTanh)
{
  v16bf aX[8], aY[8];
#pragma unroll
  for (int k = 0; k < 8; ++k) {
    aX[k] = lds_a_frag(bufX, rowA, k * 32, hi);
    aY[k] = lds_a_frag(bufY, rowA, k * 32, hi);
  }
#pragma unroll 1
  for (int nt = 0; nt < 8; ++nt) {
    int n = (ntbase + nt) * 16 + (l & 15);
    float bv = bg[n];
    v8f acc = {bv, bv, bv, bv, bv, bv, bv, bv};
#pragma unroll
    for (int k = 0; k < 8; ++k)
      acc = wmma_bf16(aX[k], wt_b_frag(Wgt, 512, n, k * 32, hi), acc);
#pragma unroll
    for (int k = 0; k < 8; ++k)
      acc = wmma_bf16(aY[k], wt_b_frag(Wgt, 512, n, 256 + k * 32, hi), acc);
#pragma unroll
    for (int r = 0; r < 8; ++r) {
      int m = rstrip + r + 8 * hi;
      float g  = 1.f / (1.f + __expf(-acc[r]));
      float xv = (float)bufX[m * RS + n];
      float hv = (float)bufY[m * RS + n];
      float v  = g * hv + (1.f - g) * xv;
      float o  = isTanh ? (1.f - 2.f / (1.f + __expf(2.f * v)))
                        : (1.f / (1.f + __expf(-v)));
      bufDst[m * RS + n] = (__bf16)o;
    }
  }
}

// ---------------- kernels ----------------

// Convert W1..W6 (256x256) and Wg (512x256) fp32 -> transposed bf16 in d_ws.
__global__ void prep_weights_kernel(const float* __restrict__ W1, const float* __restrict__ W2,
                                    const float* __restrict__ W3, const float* __restrict__ W4,
                                    const float* __restrict__ W5, const float* __restrict__ W6,
                                    const float* __restrict__ Wg, __bf16* __restrict__ wts)
{
  int gid = blockIdx.x * 256 + threadIdx.x;
  if (gid < 6 * 65536) {
    int wi = gid >> 16, e = gid & 65535, n = e >> 8, k = e & 255;
    const float* W = (wi == 0) ? W1 : (wi == 1) ? W2 : (wi == 2) ? W3
                   : (wi == 3) ? W4 : (wi == 4) ? W5 : W6;
    wts[gid] = (__bf16)W[k * 256 + n];
  } else if (gid < 6 * 65536 + 131072) {
    int e = gid - 6 * 65536, n = e >> 9, k = e & 511;
    wts[gid] = (__bf16)Wg[k * 256 + n];
  }
}

// obj_f = obj_embed[obj], obj_b = sub_embed[obj]  (pure gathers, sections 2 and 3 of d_out)
__global__ void gather_obj_kernel(const int* __restrict__ seqs,
                                  const float* __restrict__ sub_embed,
                                  const float* __restrict__ obj_embed,
                                  float* __restrict__ out)
{
  const size_t S4 = 3ull * BSZ * (H / 4);
  size_t gid = (size_t)blockIdx.x * blockDim.x + threadIdx.x;
  if (gid >= 2 * S4) return;
  int sec = (int)(gid / S4);
  size_t r = gid % S4;
  size_t row = r / (H / 4);      // t*B + b
  int c4 = (int)(r % (H / 4));
  int t = (int)(row / BSZ);
  int b = (int)(row % BSZ);
  int idx = seqs[b * 7 + 2 + 2 * t];
  const float* tab = sec ? sub_embed : obj_embed;
  float4 v = *(const float4*)(tab + (size_t)idx * H + c4 * 4);
  *(float4*)(out + (2ull + sec) * (3ull * BSZ * H) + row * (size_t)H + c4 * 4) = v;
}

__global__ void __launch_bounds__(256, 1)
rnn_scan_kernel(const int* __restrict__ seqs,
                const float* __restrict__ sub_embed,
                const float* __restrict__ rel_embed,
                const float* __restrict__ obj_embed,
                const float* __restrict__ b2, const float* __restrict__ b4,
                const float* __restrict__ b6, const float* __restrict__ bg,
                const __bf16* __restrict__ wts,
                float* __restrict__ out)
{
  extern __shared__ __bf16 lds[];
  __bf16* bufH  = lds;               // h carry; stage D writes op2 here (carry swap is free)
  __bf16* bufX  = lds + 1 * MT * RS; // x1 / x3
  __bf16* bufY  = lds + 2 * MT * RS; // h2 / h4
  __bf16* bufO1 = lds + 3 * MT * RS; // op1

  const __bf16* W1t = wts;
  const __bf16* W2t = wts + 1 * 65536;
  const __bf16* W3t = wts + 2 * 65536;
  const __bf16* W4t = wts + 3 * 65536;
  const __bf16* W5t = wts + 4 * 65536;
  const __bf16* W6t = wts + 5 * 65536;
  const __bf16* Wgt = wts + 6 * 65536;

  const int dir = blockIdx.y;
  const int b0  = blockIdx.x * MT;
  const float* stTab = dir ? obj_embed : sub_embed;
  float* outD = out + (size_t)dir * (3ull * BSZ * H);

  const int tid = threadIdx.x;
  const int w   = tid >> 5;
  const int l   = tid & 31;
  const int rstrip = (w & 3) * 16;   // wave's 16-row strip within the 64-row tile
  const int ntbase = (w >> 2) * 8;   // wave's 8 N-tiles (128-column half)
  const int hi  = l >> 4;
  const int rowA = rstrip + (l & 15);
  const int bRow = b0 + rowA;

  // init h0 = stTab[seqs[b][0]]  (cooperative fp32 -> bf16 fill)
  {
    int r = tid >> 2;
    int seg = (tid & 3) * 64;
    int idx = seqs[(b0 + r) * 7];
    const float* src = stTab + (size_t)idx * H + seg;
    __bf16* dst = bufH + r * RS + seg;
#pragma unroll
    for (int c = 0; c < 64; c += 4) {
      float4 f = *(const float4*)(src + c);
      dst[c + 0] = (__bf16)f.x; dst[c + 1] = (__bf16)f.y;
      dst[c + 2] = (__bf16)f.z; dst[c + 3] = (__bf16)f.w;
    }
  }
  __syncthreads();

#pragma unroll 1
  for (int t = 0; t < 3; ++t) {
    const int sIdx = seqs[bRow * 7 + 2 * t];
    int rIdx = seqs[bRow * 7 + 2 * t + 1];
    if (dir) { rIdx += 1000; if (rIdx >= 2000) rIdx -= 2000; }

    // Stage A1: x1 = st@W1 -> bufX ; h2 = h@W2 + b2 -> bufY
    {
      v16bf aS[8], aH[8];
#pragma unroll
      for (int k = 0; k < 8; ++k) {
        aS[k] = gather_a_frag(stTab, sIdx, k * 32, hi);
        aH[k] = lds_a_frag(bufH, rowA, k * 32, hi);
      }
      dual_gemm_to_lds(aS, aH, W1t, W2t, b2, bufX, bufY, rstrip, ntbase, l, hi);
    }
    __syncthreads();

    // Stage B: op1 = tanh(gate(x1, h2)) -> bufO1
    gate_stage(bufX, bufY, bufO1, Wgt, bg, rowA, rstrip, ntbase, l, hi, true);
    __syncthreads();

    // Stage A2: x3 = h@W3 -> bufX ; h4 = rt@W4 + b4 -> bufY
    {
      v16bf aH[8], aR[8];
#pragma unroll
      for (int k = 0; k < 8; ++k) {
        aH[k] = lds_a_frag(bufH, rowA, k * 32, hi);
        aR[k] = gather_a_frag(rel_embed, rIdx, k * 32, hi);
      }
      dual_gemm_to_lds(aH, aR, W3t, W4t, b4, bufX, bufY, rstrip, ntbase, l, hi);
    }
    __syncthreads();

    // Stage D: op2 = sigmoid(gate(x3, h4)) -> bufH (becomes next h)
    gate_stage(bufX, bufY, bufH, Wgt, bg, rowA, rstrip, ntbase, l, hi, false);
    __syncthreads();

    // Stage E: op3 = (op1@W5) * (op2@W6 + b6) -> global
    {
      v16bf a1[8], a2[8];
#pragma unroll
      for (int k = 0; k < 8; ++k) {
        a1[k] = lds_a_frag(bufO1, rowA, k * 32, hi);
        a2[k] = lds_a_frag(bufH,  rowA, k * 32, hi);
      }
#pragma unroll 1
      for (int nt = 0; nt < 8; ++nt) {
        int n = (ntbase + nt) * 16 + (l & 15);
        v8f acc5 = {};
        float bv = b6[n];
        v8f acc6 = {bv, bv, bv, bv, bv, bv, bv, bv};
#pragma unroll
        for (int k = 0; k < 8; ++k) {
          acc5 = wmma_bf16(a1[k], wt_b_frag(W5t, 256, n, k * 32, hi), acc5);
          acc6 = wmma_bf16(a2[k], wt_b_frag(W6t, 256, n, k * 32, hi), acc6);
        }
#pragma unroll
        for (int r = 0; r < 8; ++r) {
          int m = rstrip + r + 8 * hi;
          outD[((size_t)t * BSZ + b0 + m) * H + n] = acc5[r] * acc6[r];
        }
      }
    }
    __syncthreads();
  }
}

// ---------------- host entry ----------------

extern "C" void kernel_launch(void* const* d_in, const int* in_sizes, int n_in,
                              void* d_out, int out_size, void* d_ws, size_t ws_size,
                              hipStream_t stream) {
  (void)in_sizes; (void)n_in; (void)out_size; (void)ws_size;
  const int*   seqs      = (const int*)d_in[0];
  const float* sub_embed = (const float*)d_in[1];
  const float* rel_embed = (const float*)d_in[2];
  const float* obj_embed = (const float*)d_in[3];
  const float* W1 = (const float*)d_in[4];
  const float* W2 = (const float*)d_in[5];
  const float* b2 = (const float*)d_in[6];
  const float* W3 = (const float*)d_in[7];
  const float* W4 = (const float*)d_in[8];
  const float* b4 = (const float*)d_in[9];
  const float* W5 = (const float*)d_in[10];
  const float* W6 = (const float*)d_in[11];
  const float* b6 = (const float*)d_in[12];
  const float* Wg = (const float*)d_in[13];
  const float* bg = (const float*)d_in[14];
  __bf16* wts = (__bf16*)d_ws;
  float* out = (float*)d_out;

  // 1) bf16-transposed weights into workspace (524288 elements, 1 MB)
  prep_weights_kernel<<<2048, 256, 0, stream>>>(W1, W2, W3, W4, W5, W6, Wg, wts);

  // 2) obj_f / obj_b gather sections (2*3*B*H floats, as float4)
  gather_obj_kernel<<<12288, 256, 0, stream>>>(seqs, sub_embed, obj_embed, out);

  // 3) fused WMMA scan over both directions
  size_t shmem = (size_t)4 * MT * RS * sizeof(__bf16);  // 135,168 B dynamic LDS
  (void)hipFuncSetAttribute((const void*)rnn_scan_kernel,
                            hipFuncAttributeMaxDynamicSharedMemorySize, (int)shmem);
  rnn_scan_kernel<<<dim3(BSZ / MT, 2), 256, shmem, stream>>>(
      seqs, sub_embed, rel_embed, obj_embed, b2, b4, b6, bg, wts, out);
}
